// Attention3D_26439818674512
// MI455X (gfx1250) — compile-verified, matrix operand order checked by reading, not run on hardware
//
#include <hip/hip_runtime.h>
#include <hip/hip_bf16.h>
#include <stdint.h>

// Problem constants: INPUT (16,32,32), WINDOW (4,8,8), EMBED 512, HEADS 16,
// BATCH 2  ->  NW=64, M=256, N=16384, HEAD_DIM=32, SHIFT=(2,4,4),
// rows total = B*N = 32768 = 128 windows * 256 tokens.

typedef __attribute__((ext_vector_type(8)))  __bf16 v8bf;
typedef __attribute__((ext_vector_type(16))) __bf16 v16bf;
typedef __attribute__((ext_vector_type(8)))  float  v8f;

// ---------------- workspace layout (bytes) ----------------
static constexpr size_t OFF_WQ   = 0;           // W_qkv^T bf16 [1536][512] = 1572864
static constexpr size_t OFF_WP   = 1572864;     // W_proj^T bf16 [512][512] = 524288
static constexpr size_t OFF_BIAS = 2097152;     // bias f32 [16][256][256] = 4194304
static constexpr size_t OFF_Q    = 6291456;     // Q bf16 [128][16][256][32] = 33554432
static constexpr size_t OFF_K    = 39845888;    // K bf16 same
static constexpr size_t OFF_VT   = 73400320;    // V bf16 transposed [128][16][32][256]
static constexpr size_t OFF_O    = 106954752;   // attn out bf16 [128][256][512]
// total = 140509184 bytes (~134 MB) of d_ws

// fp32 -> bf16 RNE; lowers to native v_cvt_pk_bf16_f32 pairs on gfx1250.
__device__ __forceinline__ __bf16 f2bf(float f) { return (__bf16)f; }

__device__ __forceinline__ v8f zero8() {
  v8f z;
  #pragma unroll
  for (int i = 0; i < 8; ++i) z[i] = 0.0f;
  return z;
}

// Operand fragment load (A: lane = row, B: lane = column).
// Lane's 16 bf16 = two 8-elem chunks: K[hi*8..+8) and K[16+hi*8..+8), hi = lane>=16.
// 'p' points at the 32 contiguous K-values of this lane's row/column.
__device__ __forceinline__ v16bf load_frag(const __bf16* p, int hi) {
  union { v16bf v; v8bf h[2]; } u;
  u.h[0] = *(const v8bf*)(p + hi * 8);
  u.h[1] = *(const v8bf*)(p + 16 + hi * 8);
  return u.v;
}

__device__ __forceinline__ v8f wmma_bf16(v16bf a, v16bf b, v8f c) {
  return __builtin_amdgcn_wmma_f32_16x16x32_bf16(false, a, false, b, (short)0, c,
                                                 false, false);
}

// window-order row R (0..32767) -> row index in the unshifted (B,N) tensor.
// Encodes: cyclic shift by -(2,4,4) + window partition (gather), and equally the
// reverse shift + window merge (scatter), since roll(-s) gather == roll(+s) scatter.
__device__ __forceinline__ int spatial_row(int R) {
  int b   = R >> 14;
  int rem = R & 16383;
  int nw  = rem >> 8, t = rem & 255;
  int wd = nw >> 4, wh = (nw >> 2) & 3, ww = nw & 3;
  int td = t >> 6,  th = (t >> 3) & 7, tw = t & 7;
  int d = (wd * 4 + td + 2) & 15;
  int h = (wh * 8 + th + 4) & 31;
  int w = (ww * 8 + tw + 4) & 31;
  return (b << 14) + (d << 10) + (h << 5) + w;
}

// Swin shift-mask region id for token t of window nw (post-shift frame)
__device__ __forceinline__ int region_id(int nw, int t) {
  int wd = nw >> 4, wh = (nw >> 2) & 3, ww = nw & 3;
  int td = t >> 6,  th = (t >> 3) & 7, tw = t & 7;
  int ds = wd * 4 + td;    // 0..15
  int hs = wh * 8 + th;    // 0..31
  int ws = ww * 8 + tw;    // 0..31
  int rd = (ds < 12) ? 0 : ((ds < 14) ? 1 : 2);
  int rh = (hs < 24) ? 0 : ((hs < 28) ? 1 : 2);
  int rw = (ws < 24) ? 0 : ((ws < 28) ? 1 : 2);
  return rd * 9 + rh * 3 + rw;
}

// ---------------- prep kernels ----------------
// Convert weights to bf16 AND transpose to column-major [n][k], so the GEMM
// B-tile staging is a contiguous copy (no per-element LDS scatter in the loop).
__global__ __launch_bounds__(256) void prep_weights(const float* __restrict__ wq,
                                                    const float* __restrict__ wp,
                                                    __bf16* __restrict__ wqT,
                                                    __bf16* __restrict__ wpT) {
  int i = blockIdx.x * 256 + threadIdx.x;      // grid covers 512*1536
  if (i < 512 * 1536) {
    int k = i / 1536, n = i % 1536;            // coalesced read over n
    wqT[(size_t)n * 512 + k] = f2bf(wq[i]);
  }
  if (i < 512 * 512) {
    int k = i >> 9, n = i & 511;
    wpT[(size_t)n * 512 + k] = f2bf(wp[i]);
  }
}

// bias[h][i][j] = sum over (d,h,w) components of bias_table[idx][h]
__global__ __launch_bounds__(256) void prep_bias(const float* __restrict__ table,
                                                 float* __restrict__ bias) {
  int id = blockIdx.x * 256 + threadIdx.x;     // < 16*256*256
  int h = id >> 16, rem = id & 65535;
  int i = rem >> 8, j = rem & 255;
  int idl = i >> 6, ihl = (i >> 3) & 7, iwl = i & 7;
  int jdl = j >> 6, jhl = (j >> 3) & 7, jwl = j & 7;
  int ia = (jdl - idl) + 3;            // 0..6
  int ib = (jhl - ihl) + 14;           // 7..21
  int ic = (jwl - iwl) + 29;           // 22..36
  bias[id] = table[ia * 16 + h] + table[ib * 16 + h] + table[ic * 16 + h];
}

// ---------------- kernel 1: fused shift+window gather + QKV GEMM ----------------
// C[32768 x 1536] = gather(x)[32768 x 512] (bf16) @ Wqkv (bf16), f32 accum.
// 128x128 tile / workgroup, 8 waves (4 row-groups x 2 col-groups), k-step 32.
__global__ __launch_bounds__(256) void qkv_gemm(const float* __restrict__ x,
                                                const __bf16* __restrict__ wqT,
                                                __bf16* __restrict__ Qb,
                                                __bf16* __restrict__ Kb,
                                                __bf16* __restrict__ Vt) {
  __shared__ __align__(16) __bf16 aT[128 * 40];   // [row][k] padded 32->40
  __shared__ __align__(16) __bf16 bT[128 * 40];   // [col][k] padded
  const int tid  = threadIdx.x;
  const int lane = tid & 31, w = tid >> 5;
  const int wrow = w >> 1, wcol = w & 1;
  const int hi = lane >> 4, ln = lane & 15;
  const int rb = blockIdx.x * 128;
  const int nb = blockIdx.y * 128;

  // fixed staging assignment per thread: line li (row or col), 16-elem half lh
  const int li = tid >> 1, lh = tid & 1;
  const int srcRow = spatial_row(rb + li);        // fused shift+window gather
  const float*  aSrc = x + (size_t)srcRow * 512 + lh * 16;
  const __bf16* bSrc = wqT + (size_t)(nb + li) * 512 + lh * 16;  // pre-transposed
  __bf16* aDst = aT + li * 40 + lh * 16;
  __bf16* bDst = bT + li * 40 + lh * 16;

  v8f acc[2][4];
  #pragma unroll
  for (int i = 0; i < 2; ++i)
    #pragma unroll
    for (int j = 0; j < 4; ++j) acc[i][j] = zero8();

  for (int k0 = 0; k0 < 512; k0 += 32) {
    // stage A (fp32 -> bf16 convert, contiguous)
    {
      const float4* p = (const float4*)(aSrc + k0);
      float4 f0 = p[0], f1 = p[1], f2 = p[2], f3 = p[3];
      union { v8bf v; __bf16 e[8]; } c0, c1;
      c0.e[0]=f2bf(f0.x); c0.e[1]=f2bf(f0.y); c0.e[2]=f2bf(f0.z); c0.e[3]=f2bf(f0.w);
      c0.e[4]=f2bf(f1.x); c0.e[5]=f2bf(f1.y); c0.e[6]=f2bf(f1.z); c0.e[7]=f2bf(f1.w);
      c1.e[0]=f2bf(f2.x); c1.e[1]=f2bf(f2.y); c1.e[2]=f2bf(f2.z); c1.e[3]=f2bf(f2.w);
      c1.e[4]=f2bf(f3.x); c1.e[5]=f2bf(f3.y); c1.e[6]=f2bf(f3.z); c1.e[7]=f2bf(f3.w);
      *(v8bf*)(aDst)     = c0.v;
      *(v8bf*)(aDst + 8) = c1.v;
    }
    // stage B (bf16 contiguous copy from pre-transposed weights)
    {
      const uint4* p = (const uint4*)(bSrc + k0);
      uint4 b0 = p[0], b1 = p[1];
      uint4* d = (uint4*)bDst;
      d[0] = b0; d[1] = b1;
    }
    if (k0 + 32 < 512) {                  // prefetch next k-slice
      __builtin_prefetch(aSrc + k0 + 32, 0, 1);
      __builtin_prefetch(bSrc + k0 + 32, 0, 1);
    }
    __syncthreads();

    v16bf af[2], bfr[4];
    #pragma unroll
    for (int rt = 0; rt < 2; ++rt)
      af[rt] = load_frag(aT + (wrow * 32 + rt * 16 + ln) * 40, hi);
    #pragma unroll
    for (int ct = 0; ct < 4; ++ct)
      bfr[ct] = load_frag(bT + (wcol * 64 + ct * 16 + ln) * 40, hi);
    #pragma unroll
    for (int rt = 0; rt < 2; ++rt)
      #pragma unroll
      for (int ct = 0; ct < 4; ++ct)
        acc[rt][ct] = wmma_bf16(af[rt], bfr[ct], acc[rt][ct]);
    __syncthreads();
  }

  // epilogue: split into Q/K row-major and V transposed, bf16
  #pragma unroll
  for (int rt = 0; rt < 2; ++rt) {
    #pragma unroll
    for (int ct = 0; ct < 4; ++ct) {
      int c0   = nb + wcol * 64 + ct * 16;
      int mat  = c0 >> 9;               // 0=Q 1=K 2=V (uniform per tile)
      int head = (c0 >> 5) & 15;        // uniform per tile
      int chn  = (c0 & 31) + ln;
      int R0   = rb + wrow * 32 + rt * 16 + hi * 8;
      #pragma unroll
      for (int r = 0; r < 8; ++r) {
        int R = R0 + r;
        int win = R >> 8, tok = R & 255;
        __bf16 val = f2bf(acc[rt][ct][r]);
        if (mat == 0)
          Qb[((size_t)(win * 16 + head) * 256 + tok) * 32 + chn] = val;
        else if (mat == 1)
          Kb[((size_t)(win * 16 + head) * 256 + tok) * 32 + chn] = val;
        else
          Vt[((size_t)(win * 16 + head) * 32 + chn) * 256 + tok] = val;
      }
    }
  }
}

// ---------------- kernel 2: attention, one workgroup per (window, head) ----------------
// Full 256x256 score matrix resident in LDS (CDNA5 320KB WGP LDS).
// LDS: S f32 [256][260] (266240 B) | K bf16 [256][40] (20480 B) |
//      Vt bf16 [32][264] (16896 B) | region u8 [256]
static constexpr unsigned ATTN_LDS = 266240u + 20480u + 16896u + 256u; // 303872

__global__ __launch_bounds__(256) void attn(const __bf16* __restrict__ Qb,
                                            const __bf16* __restrict__ Kb,
                                            const __bf16* __restrict__ Vt,
                                            const float* __restrict__ biasb,
                                            __bf16* __restrict__ Ob) {
  extern __shared__ __align__(16) char smem[];
  float*   sS   = (float*)smem;                       // [256][260]
  __bf16*  sK   = (__bf16*)(smem + 266240);           // [256][40]
  __bf16*  sV   = (__bf16*)(smem + 286720);           // [32][264]
  uint8_t* sReg = (uint8_t*)(smem + 303616);          // [256] region ids

  const int tid = threadIdx.x, lane = tid & 31, w = tid >> 5;
  const int hi = lane >> 4, ln = lane & 15;
  const int win = blockIdx.x >> 4, head = blockIdx.x & 15;
  const int nw = win & 63;
  const size_t whQK = (size_t)(win * 16 + head) * 256 * 32;
  const size_t whV  = (size_t)(win * 16 + head) * 32 * 256;

  // region-id table: depends only on (nw, token); one thread per token
  sReg[tid] = (uint8_t)region_id(nw, tid);

  // stage K rows (each thread copies one key row, 64B)
  {
    const uint4* s = (const uint4*)(Kb + whQK + (size_t)tid * 32);
    uint4 a0 = s[0], a1 = s[1], a2 = s[2], a3 = s[3];
    uint4* d = (uint4*)(sK + tid * 40);
    d[0] = a0; d[1] = a1; d[2] = a2; d[3] = a3;
  }
  // stage V (already transposed [32 ch][256 keys] in global)
  {
    int vr = tid >> 3, vs = (tid & 7) * 32;
    const uint4* s = (const uint4*)(Vt + whV + (size_t)vr * 256 + vs);
    uint4 a0 = s[0], a1 = s[1], a2 = s[2], a3 = s[3];
    uint4* d = (uint4*)(sV + vr * 264 + vs);
    d[0] = a0; d[1] = a1; d[2] = a2; d[3] = a3;
  }
  // Q fragments straight from global (A-operand layout, head_dim = K = 32)
  v16bf qf[2];
  #pragma unroll
  for (int rt = 0; rt < 2; ++rt)
    qf[rt] = load_frag(Qb + whQK + (size_t)(w * 32 + rt * 16 + ln) * 32, hi);
  __syncthreads();

  // phase 1: raw S = Q K^T into LDS (each wave owns 32 query rows)
  for (int jt = 0; jt < 16; ++jt) {
    v16bf kf = load_frag(sK + (jt * 16 + ln) * 40, hi);
    #pragma unroll
    for (int rt = 0; rt < 2; ++rt) {
      v8f acc = zero8();
      acc = wmma_bf16(qf[rt], kf, acc);
      int i0 = w * 32 + rt * 16 + hi * 8;
      #pragma unroll
      for (int r = 0; r < 8; ++r)
        sS[(i0 + r) * 260 + jt * 16 + ln] = acc[r];
    }
  }
  __syncthreads();

  // softmax: one thread per query row.  s = (qk + bias)*scale + mask
  {
    const int i = tid;
    float* row = sS + (size_t)i * 260;
    const float* brow = biasb + (size_t)head * 65536 + (size_t)i * 256;
    const int ri = sReg[i];
    float mx = -3.0e38f;
    for (int j4 = 0; j4 < 256; j4 += 4) {          // float4-vectorized pass
      float4 s4 = *(float4*)(row + j4);
      float4 b4 = *(const float4*)(brow + j4);
      float v0 = (s4.x + b4.x) * 0.17677669529663687f;   // 1/sqrt(32)
      float v1 = (s4.y + b4.y) * 0.17677669529663687f;
      float v2 = (s4.z + b4.z) * 0.17677669529663687f;
      float v3 = (s4.w + b4.w) * 0.17677669529663687f;
      if (sReg[j4 + 0] != ri) v0 -= 1.0e9f;              // shift mask
      if (sReg[j4 + 1] != ri) v1 -= 1.0e9f;
      if (sReg[j4 + 2] != ri) v2 -= 1.0e9f;
      if (sReg[j4 + 3] != ri) v3 -= 1.0e9f;
      *(float4*)(row + j4) = make_float4(v0, v1, v2, v3);
      mx = fmaxf(mx, fmaxf(fmaxf(v0, v1), fmaxf(v2, v3)));
    }
    float sum = 0.0f;
    for (int j4 = 0; j4 < 256; j4 += 4) {
      float4 s4 = *(float4*)(row + j4);
      float e0 = __expf(s4.x - mx), e1 = __expf(s4.y - mx);
      float e2 = __expf(s4.z - mx), e3 = __expf(s4.w - mx);
      *(float4*)(row + j4) = make_float4(e0, e1, e2, e3);
      sum += (e0 + e1) + (e2 + e3);
    }
    float inv = 1.0f / sum;
    __bf16* prow = (__bf16*)row;       // in-place bf16 pack (thread owns row)
    for (int j4 = 0; j4 < 256; j4 += 4) {
      float4 s4 = *(float4*)(row + j4);
      prow[j4 + 0] = f2bf(s4.x * inv);
      prow[j4 + 1] = f2bf(s4.y * inv);
      prow[j4 + 2] = f2bf(s4.z * inv);
      prow[j4 + 3] = f2bf(s4.w * inv);
    }
  }
  __syncthreads();

  // phase 2: O = P V   (K-dim = 256 keys, 8 steps of 32)
  v8f oa[2][2];
  #pragma unroll
  for (int rt = 0; rt < 2; ++rt)
    #pragma unroll
    for (int ct = 0; ct < 2; ++ct) oa[rt][ct] = zero8();

  for (int k8 = 0; k8 < 8; ++k8) {
    v16bf pf[2], vf[2];
    #pragma unroll
    for (int rt = 0; rt < 2; ++rt)
      pf[rt] = load_frag((const __bf16*)((const char*)sS +
                         (size_t)(w * 32 + rt * 16 + ln) * 1040) + k8 * 32, hi);
    #pragma unroll
    for (int ct = 0; ct < 2; ++ct)
      vf[ct] = load_frag(sV + (ct * 16 + ln) * 264 + k8 * 32, hi);
    #pragma unroll
    for (int rt = 0; rt < 2; ++rt)
      #pragma unroll
      for (int ct = 0; ct < 2; ++ct)
        oa[rt][ct] = wmma_bf16(pf[rt], vf[ct], oa[rt][ct]);
  }

  // epilogue: merged-head bf16 output [win][tok][512]
  #pragma unroll
  for (int rt = 0; rt < 2; ++rt) {
    #pragma unroll
    for (int ct = 0; ct < 2; ++ct) {
      int m0 = w * 32 + rt * 16 + hi * 8;
      int c  = head * 32 + ct * 16 + ln;
      #pragma unroll
      for (int r = 0; r < 8; ++r)
        Ob[(size_t)(win * 256 + m0 + r) * 512 + c] = f2bf(oa[rt][ct][r]);
    }
  }
}

// ---------------- kernel 3: output projection + fused reverse shift scatter ----------------
__global__ __launch_bounds__(256) void proj_gemm(const __bf16* __restrict__ Ob,
                                                 const __bf16* __restrict__ wpT,
                                                 const float* __restrict__ bproj,
                                                 float* __restrict__ out) {
  __shared__ __align__(16) __bf16 aT[128 * 40];
  __shared__ __align__(16) __bf16 bT[128 * 40];
  const int tid  = threadIdx.x;
  const int lane = tid & 31, w = tid >> 5;
  const int wrow = w >> 1, wcol = w & 1;
  const int hi = lane >> 4, ln = lane & 15;
  const int rb = blockIdx.x * 128;
  const int nb = blockIdx.y * 128;

  const int li = tid >> 1, lh = tid & 1;
  const __bf16* aSrc = Ob + (size_t)(rb + li) * 512 + lh * 16;
  const __bf16* bSrc = wpT + (size_t)(nb + li) * 512 + lh * 16;  // pre-transposed
  __bf16* aDst = aT + li * 40 + lh * 16;
  __bf16* bDst = bT + li * 40 + lh * 16;

  v8f acc[2][4];
  #pragma unroll
  for (int i = 0; i < 2; ++i)
    #pragma unroll
    for (int j = 0; j < 4; ++j) acc[i][j] = zero8();

  for (int k0 = 0; k0 < 512; k0 += 32) {
    {
      const uint4* p = (const uint4*)(aSrc + k0);
      uint4 a0 = p[0], a1 = p[1];
      uint4* d = (uint4*)aDst;
      d[0] = a0; d[1] = a1;
    }
    {
      const uint4* p = (const uint4*)(bSrc + k0);
      uint4 b0 = p[0], b1 = p[1];
      uint4* d = (uint4*)bDst;
      d[0] = b0; d[1] = b1;
    }
    if (k0 + 32 < 512) {
      __builtin_prefetch(aSrc + k0 + 32, 0, 1);
      __builtin_prefetch(bSrc + k0 + 32, 0, 1);
    }
    __syncthreads();

    v16bf af[2], bfr[4];
    #pragma unroll
    for (int rt = 0; rt < 2; ++rt)
      af[rt] = load_frag(aT + (wrow * 32 + rt * 16 + ln) * 40, hi);
    #pragma unroll
    for (int ct = 0; ct < 4; ++ct)
      bfr[ct] = load_frag(bT + (wcol * 64 + ct * 16 + ln) * 40, hi);
    #pragma unroll
    for (int rt = 0; rt < 2; ++rt)
      #pragma unroll
      for (int ct = 0; ct < 4; ++ct)
        acc[rt][ct] = wmma_bf16(af[rt], bfr[ct], acc[rt][ct]);
    __syncthreads();
  }

  // epilogue: + b_proj, scatter to (B,N,C) with reverse window/roll mapping
  #pragma unroll
  for (int rt = 0; rt < 2; ++rt) {
    #pragma unroll
    for (int ct = 0; ct < 4; ++ct) {
      int c = nb + wcol * 64 + ct * 16 + ln;
      float bc = bproj[c];
      int R0 = rb + wrow * 32 + rt * 16 + hi * 8;
      #pragma unroll
      for (int r = 0; r < 8; ++r) {
        int dst = spatial_row(R0 + r);
        out[(size_t)dst * 512 + c] = acc[rt][ct][r] + bc;
      }
    }
  }
}

// ---------------- launcher ----------------
extern "C" void kernel_launch(void* const* d_in, const int* in_sizes, int n_in,
                              void* d_out, int out_size, void* d_ws, size_t ws_size,
                              hipStream_t stream) {
  (void)in_sizes; (void)n_in; (void)out_size; (void)ws_size;
  const float* x     = (const float*)d_in[0];
  const float* Wqkv  = (const float*)d_in[1];
  const float* Wproj = (const float*)d_in[2];
  const float* bproj = (const float*)d_in[3];
  const float* btab  = (const float*)d_in[4];

  char* ws = (char*)d_ws;
  __bf16* wqT   = (__bf16*)(ws + OFF_WQ);
  __bf16* wpT   = (__bf16*)(ws + OFF_WP);
  float*  biasb = (float*)(ws + OFF_BIAS);
  __bf16* Qb    = (__bf16*)(ws + OFF_Q);
  __bf16* Kb    = (__bf16*)(ws + OFF_K);
  __bf16* Vtb   = (__bf16*)(ws + OFF_VT);
  __bf16* Obuf  = (__bf16*)(ws + OFF_O);

  prep_weights<<<3072, 256, 0, stream>>>(Wqkv, Wproj, wqT, wpT);
  prep_bias<<<4096, 256, 0, stream>>>(btab, biasb);
  qkv_gemm<<<dim3(256, 12), 256, 0, stream>>>(x, wqT, Qb, Kb, Vtb);
  attn<<<2048, 256, ATTN_LDS, stream>>>(Qb, Kb, Vtb, biasb, Obuf);
  proj_gemm<<<dim3(256, 4), 256, 0, stream>>>(Obuf, wpT, bproj, (float*)d_out);
}
